// PretrainedSPPPViTWithMHLA_72009421685105
// MI455X (gfx1250) — compile-verified, matrix operand order checked by reading, not run on hardware
//
#include <hip/hip_runtime.h>
#include <hip/hip_bf16.h>

typedef __attribute__((ext_vector_type(16))) _Float16 v16h;
typedef __attribute__((ext_vector_type(8)))  _Float16 v8h;
typedef __attribute__((ext_vector_type(8)))  float    v8f;

namespace vitcfg {
constexpr int BATCH = 64, IMG = 224, HP = 14, NPATCH = 196, RSEG = 196;
constexpr int DIM = 384, HEADS = 6, HD = 64, DEPTH = 12, MLPD = 1536, CLSN = 1000;
constexpr int SEQ = 197, WIN = 4;
constexpr int MTOK = BATCH * SEQ;      // 12608
constexpr int MPATCH = BATCH * NPATCH; // 12544
constexpr int PKDIM = 768;             // 3*16*16
constexpr int QKVD = 3 * DIM;          // 1152
constexpr int HEADPAD = 1024;          // 1000 padded so Ntiles % 4 == 0
constexpr int PIX = IMG * IMG;
}
using namespace vitcfg;

union H16pack { v16h v; v8h h[2]; };

__device__ __forceinline__ float gelu_approx(float x) {
  float x3 = x * x * x;
  return 0.5f * x * (1.f + tanhf(0.7978845608028654f * (x + 0.044715f * x3)));
}

// ---------------------------------------------------------------------------
// Register-blocked WMMA GEMM: each wave computes a 32(M) x 64(N) macro-tile
// as 2x4 = 8 independent 16x16 WMMA accumulators (v_wmma_f32_16x16x32_f16,
// f32 accumulate). Per K-step of 32: 12 b128 loads feed 8 WMMAs (~21 FLOP/B
// from cache vs 8 for the unblocked version), and 8 independent WMMAs hide
// the WMMA->WMMA dependency latency.
// Fragment layout per CDNA5 ISA 7.12.2:
//  A 16x32: lanes 0-15 row M=lane, K{kk..kk+7, kk+16..kk+23};
//           lanes 16-31 row M=lane-16, K{kk+8..kk+15, kk+24..kk+31}.
//  B mirrored by column (Bt stored [N][K] so loads are contiguous in K).
//  C/D: VGPR r -> M = r + 8*(lane>=16), N = lane&15.
// Requires M % 32 == 0 and Ntiles % 4 == 0 (pad weights; stores guarded by
// Nreal).
// ---------------------------------------------------------------------------
template <bool GELU_, bool RES, bool WF32, bool WF16>
__global__ __launch_bounds__(256) void gemm_wmma(
    const _Float16* __restrict__ A, const _Float16* __restrict__ Bt,
    const float* __restrict__ bias, const float* __restrict__ resid,
    float* __restrict__ out32, _Float16* __restrict__ out16,
    int M, int Ntiles, int Nreal, int K) {
  int wave = (blockIdx.x * blockDim.x + threadIdx.x) >> 5;
  int lane = threadIdx.x & 31;
  int mt2 = M >> 5;        // 32-row supertiles
  int nt4 = Ntiles >> 2;   // 64-col supertiles
  if (wave >= mt2 * nt4) return;
  int tn = wave % nt4, tm = wave / nt4;
  int hlf = lane >> 4;     // which K-half this lane holds
  int l16 = lane & 15;
  const _Float16* ap = A + (size_t)(tm * 32 + l16) * K + hlf * 8;
  const _Float16* bp = Bt + (size_t)(tn * 64 + l16) * K + hlf * 8;
  const size_t tstride = (size_t)16 * K;  // one 16-row tile step in A or Bt
  v8f acc[2][4];
#pragma unroll
  for (int i = 0; i < 2; ++i)
#pragma unroll
    for (int j = 0; j < 4; ++j) acc[i][j] = (v8f){};

  for (int kk = 0; kk < K; kk += 32) {
    H16pack a[2], b[4];
#pragma unroll
    for (int i = 0; i < 2; ++i) {
      a[i].h[0] = *(const v8h*)(ap + i * tstride + kk);
      a[i].h[1] = *(const v8h*)(ap + i * tstride + kk + 16);
    }
#pragma unroll
    for (int j = 0; j < 4; ++j) {
      b[j].h[0] = *(const v8h*)(bp + j * tstride + kk);
      b[j].h[1] = *(const v8h*)(bp + j * tstride + kk + 16);
    }
#pragma unroll
    for (int j = 0; j < 4; ++j)
#pragma unroll
      for (int i = 0; i < 2; ++i)
        acc[i][j] = __builtin_amdgcn_wmma_f32_16x16x32_f16(
            false, a[i].v, false, b[j].v, (short)0, acc[i][j], false, false);
  }

#pragma unroll
  for (int j = 0; j < 4; ++j) {
    int n = tn * 64 + j * 16 + l16;
    if (n < Nreal) {
      float bs = bias ? bias[n] : 0.f;
#pragma unroll
      for (int i = 0; i < 2; ++i) {
        int mbase = tm * 32 + i * 16 + hlf * 8;
#pragma unroll
        for (int r = 0; r < 8; ++r) {
          float v = acc[i][j][r] + bs;
          if (GELU_) v = gelu_approx(v);
          size_t idx = (size_t)(mbase + r) * Nreal + n;
          if (RES) v += resid[idx];
          if (WF32) out32[idx] = v;
          if (WF16) out16[idx] = (_Float16)v;
        }
      }
    }
  }
}

// f32 [K][N] row-major -> f16 [Npad][K] (transposed, pad rows zeroed)
__global__ __launch_bounds__(256) void conv_transpose_f16(
    const float* __restrict__ src, _Float16* __restrict__ dst,
    int K, int N, int Npad) {
  int t = blockIdx.x * blockDim.x + threadIdx.x;
  int tot = Npad * K;
  if (t >= tot) return;
  int k = t % K, n = t / K;
  dst[t] = (n < N) ? (_Float16)src[(size_t)k * N + n] : (_Float16)0.f;
}

__global__ __launch_bounds__(256) void zero_f32(float* p, int n) {
  int t = blockIdx.x * blockDim.x + threadIdx.x;
  if (t < n) p[t] = 0.f;
}

// Gather x (B,3,224,224) into patch rows xp16[B*196][768] (f16)
__global__ __launch_bounds__(256) void patch_gather(
    const float* __restrict__ x, _Float16* __restrict__ xp16) {
  int t = blockIdx.x * blockDim.x + threadIdx.x;
  if (t >= MPATCH * PKDIM) return;
  int f = t % PKDIM, row = t / PKDIM;
  int n = row % NPATCH, b = row / NPATCH;
  int c = f >> 8, py = (f >> 4) & 15, px = f & 15;
  int hy = n / HP, hx = n % HP;
  int iy = hy * 16 + py, ix = hx * 16 + px;
  xp16[t] = (_Float16)x[((size_t)(b * 3 + c) * IMG + iy) * IMG + ix];
}

// Segment-sum patch embeddings into psum[B*R][D] via center-pixel seg id
__global__ __launch_bounds__(256) void pool_accum(
    const float* __restrict__ emb, const int* __restrict__ seg,
    float* __restrict__ psum) {
  int t = blockIdx.x * blockDim.x + threadIdx.x;
  if (t >= MPATCH * DIM) return;
  int d = t % DIM, row = t / DIM;
  int n = row % NPATCH, b = row / NPATCH;
  int cy = (n / HP) * 16 + 8, cx = (n % HP) * 16 + 8;
  int s = seg[(size_t)b * PIX + cy * IMG + cx];
  atomicAdd(&psum[((size_t)b * RSEG + s) * DIM + d], emb[t]);
}

__global__ __launch_bounds__(256) void pool_count(
    const int* __restrict__ seg, float* __restrict__ cnt) {
  int row = blockIdx.x * blockDim.x + threadIdx.x;
  if (row >= MPATCH) return;
  int n = row % NPATCH, b = row / NPATCH;
  int cy = (n / HP) * 16 + 8, cx = (n % HP) * 16 + 8;
  int s = seg[(size_t)b * PIX + cy * IMG + cx];
  atomicAdd(&cnt[b * RSEG + s], 1.f);
}

// Per-pixel centroid accumulation
__global__ __launch_bounds__(256) void centroid_accum(
    const int* __restrict__ seg, float* __restrict__ pcnt,
    float* __restrict__ xsum, float* __restrict__ ysum) {
  int t = blockIdx.x * blockDim.x + threadIdx.x;
  if (t >= BATCH * PIX) return;
  int b = t / PIX, p = t % PIX;
  int i = p / IMG, j = p % IMG;
  int s = seg[t];
  int idx = b * RSEG + s;
  atomicAdd(&pcnt[idx], 1.f);
  atomicAdd(&xsum[idx], (float)j * (1.f / IMG));
  atomicAdd(&ysum[idx], (float)i * (1.f / IMG));
}

// h[b,s,:] = (s==0 ? cls+cls_pos : pooled_mean + centroid@pos_W + pos_b)
__global__ __launch_bounds__(256) void assemble_tokens(
    const float* __restrict__ psum, const float* __restrict__ cnt,
    const float* __restrict__ pcnt, const float* __restrict__ xsum,
    const float* __restrict__ ysum, const float* __restrict__ cls_tok,
    const float* __restrict__ cls_pos, const float* __restrict__ pos_W,
    const float* __restrict__ pos_b, float* __restrict__ h) {
  int t = blockIdx.x * blockDim.x + threadIdx.x;
  if (t >= MTOK * DIM) return;
  int d = t % DIM, row = t / DIM;
  int s = row % SEQ, b = row / SEQ;
  float v;
  if (s == 0) {
    v = cls_tok[d] + cls_pos[d];
  } else {
    int idx = b * RSEG + (s - 1);
    float pooled = psum[(size_t)idx * DIM + d] / fmaxf(cnt[idx], 1.f);
    float pc = pcnt[idx];
    float safe = fmaxf(pc, 1.f);
    float cx = pc > 0.f ? xsum[idx] / safe : 0.5f;
    float cy = pc > 0.f ? ysum[idx] / safe : 0.5f;
    v = pooled + cx * pos_W[d] + cy * pos_W[DIM + d] + pos_b[d];
  }
  h[t] = v;
}

// LayerNorm: one wave per row of 384, writes f16 row into out16.
__global__ __launch_bounds__(256) void ln_rows(
    const float* __restrict__ in, size_t in_stride,
    const float* __restrict__ w, const float* __restrict__ b,
    _Float16* __restrict__ out16, int rows) {
  int wave = (blockIdx.x * blockDim.x + threadIdx.x) >> 5;
  int lane = threadIdx.x & 31;
  if (wave >= rows) return;
  const float* x = in + (size_t)wave * in_stride;
  float vals[12];
  float s = 0.f;
#pragma unroll
  for (int i = 0; i < 12; ++i) { vals[i] = x[lane + i * 32]; s += vals[i]; }
#pragma unroll
  for (int o = 16; o; o >>= 1) s += __shfl_xor(s, o, 32);
  float mean = s * (1.f / DIM);
  float vv = 0.f;
#pragma unroll
  for (int i = 0; i < 12; ++i) { float d = vals[i] - mean; vv += d * d; }
#pragma unroll
  for (int o = 16; o; o >>= 1) vv += __shfl_xor(vv, o, 32);
  float rstd = rsqrtf(vv * (1.f / DIM) + 1e-5f);
#pragma unroll
  for (int i = 0; i < 12; ++i) {
    int c = lane + i * 32;
    out16[(size_t)wave * DIM + c] = (_Float16)((vals[i] - mean) * rstd * w[c] + b[c]);
  }
}

// Banded attention, one thread per (b, head, q) row, online softmax over the
// allowed key set: {0} ∪ [q-4, q+4] (row 0 attends to everything).
__global__ __launch_bounds__(256) void banded_attn(
    const float* __restrict__ qkv, _Float16* __restrict__ out16) {
  int t = blockIdx.x * blockDim.x + threadIdx.x;
  if (t >= BATCH * HEADS * SEQ) return;
  int q_ = t % SEQ;
  int bh = t / SEQ;
  int h_ = bh % HEADS, b_ = bh / HEADS;
  const float* base = qkv + (size_t)b_ * SEQ * QKVD;
  const float* qrow = base + (size_t)q_ * QKVD + h_ * HD;
  float qr[HD], o[HD];
#pragma unroll
  for (int d = 0; d < HD; ++d) { qr[d] = qrow[d] * 0.125f; o[d] = 0.f; }
  float mx = -3.0e38f, l = 0.f;
  auto attend = [&](int j) {
    const float* kr = base + (size_t)j * QKVD + DIM + h_ * HD;
    const float* vr = base + (size_t)j * QKVD + 2 * DIM + h_ * HD;
    float s = 0.f;
#pragma unroll
    for (int d = 0; d < HD; ++d) s += qr[d] * kr[d];
    float mn = fmaxf(mx, s);
    float corr = __expf(mx - mn);
    float p = __expf(s - mn);
    l = l * corr + p;
#pragma unroll
    for (int d = 0; d < HD; ++d) o[d] = o[d] * corr + p * vr[d];
    mx = mn;
  };
  if (q_ == 0) {
    for (int j = 0; j < SEQ; ++j) attend(j);
  } else {
    attend(0);
    int j0 = q_ - WIN < 1 ? 1 : q_ - WIN;
    int j1 = q_ + WIN > SEQ - 1 ? SEQ - 1 : q_ + WIN;
    for (int j = j0; j <= j1; ++j) attend(j);
  }
  float inv = 1.f / l;
  _Float16* op = out16 + (size_t)(b_ * SEQ + q_) * DIM + h_ * HD;
#pragma unroll
  for (int d = 0; d < HD; ++d) op[d] = (_Float16)(o[d] * inv);
}

// ---------------------------------------------------------------------------
extern "C" void kernel_launch(void* const* d_in, const int* in_sizes, int n_in,
                              void* d_out, int out_size, void* d_ws, size_t ws_size,
                              hipStream_t stream) {
  (void)in_sizes; (void)n_in; (void)out_size; (void)ws_size;
  const float* x       = (const float*)d_in[0];
  const int*   seg     = (const int*)d_in[1];
  const float* patch_W = (const float*)d_in[2];
  const float* patch_b = (const float*)d_in[3];
  const float* cls_tok = (const float*)d_in[4];
  const float* pos_W   = (const float*)d_in[5];
  const float* pos_b   = (const float*)d_in[6];
  const float* cls_pos = (const float*)d_in[7];
  const float* ln1_w   = (const float*)d_in[8];
  const float* ln1_b   = (const float*)d_in[9];
  const float* qkv_W   = (const float*)d_in[10];
  const float* qkv_b   = (const float*)d_in[11];
  const float* proj_W  = (const float*)d_in[12];
  const float* proj_b  = (const float*)d_in[13];
  const float* ln2_w   = (const float*)d_in[14];
  const float* ln2_b   = (const float*)d_in[15];
  const float* mlp_W1  = (const float*)d_in[16];
  const float* mlp_b1  = (const float*)d_in[17];
  const float* mlp_W2  = (const float*)d_in[18];
  const float* mlp_b2  = (const float*)d_in[19];
  const float* norm_w  = (const float*)d_in[20];
  const float* norm_b  = (const float*)d_in[21];
  const float* head_W  = (const float*)d_in[22];
  const float* head_b  = (const float*)d_in[23];
  float* out = (float*)d_out;

  char* ws = (char*)d_ws;
  size_t off = 0;
  auto alloc = [&](size_t bytes) -> char* {
    char* p = ws + off;
    off = (off + bytes + 255) & ~(size_t)255;
    return p;
  };
  _Float16* pw_t   = (_Float16*)alloc((size_t)DIM * PKDIM * 2);
  _Float16* qkv_t  = (_Float16*)alloc((size_t)DEPTH * QKVD * DIM * 2);
  _Float16* proj_t = (_Float16*)alloc((size_t)DEPTH * DIM * DIM * 2);
  _Float16* mlp1_t = (_Float16*)alloc((size_t)DEPTH * MLPD * DIM * 2);
  _Float16* mlp2_t = (_Float16*)alloc((size_t)DEPTH * DIM * MLPD * 2);
  _Float16* head_t = (_Float16*)alloc((size_t)HEADPAD * DIM * 2);
  _Float16* xp16   = (_Float16*)alloc((size_t)MPATCH * PKDIM * 2);
  float*    emb    = (float*)alloc((size_t)MPATCH * DIM * 4);
  float*    psum   = (float*)alloc((size_t)BATCH * RSEG * DIM * 4);
  float*    cnt    = (float*)alloc((size_t)BATCH * RSEG * 4);
  float*    pcnt   = (float*)alloc((size_t)BATCH * RSEG * 4);
  float*    xsum   = (float*)alloc((size_t)BATCH * RSEG * 4);
  float*    ysum   = (float*)alloc((size_t)BATCH * RSEG * 4);
  float*    h      = (float*)alloc((size_t)MTOK * DIM * 4);
  _Float16* z16    = (_Float16*)alloc((size_t)MTOK * DIM * 2);
  float*    qkvbuf = (float*)alloc((size_t)MTOK * QKVD * 4);
  _Float16* attn16 = (_Float16*)alloc((size_t)MTOK * DIM * 2);
  _Float16* mid16  = (_Float16*)alloc((size_t)MTOK * MLPD * 2);
  _Float16* cls16  = (_Float16*)alloc((size_t)BATCH * DIM * 2);

  auto cdiv = [](int a, int b) { return (a + b - 1) / b; };
  auto convT = [&](const float* src, _Float16* dst, int K, int N, int Npad) {
    int tot = Npad * K;
    conv_transpose_f16<<<cdiv(tot, 256), 256, 0, stream>>>(src, dst, K, N, Npad);
  };
  auto gtiles = [](int M, int Ntiles) { return (M >> 5) * (Ntiles >> 2); };

  // --- weight conversion (f32 -> transposed f16) ---
  convT(patch_W, pw_t, PKDIM, DIM, DIM);
  convT(head_W, head_t, DIM, CLSN, HEADPAD);
  for (int l = 0; l < DEPTH; ++l) {
    convT(qkv_W + (size_t)l * DIM * QKVD, qkv_t + (size_t)l * QKVD * DIM, DIM, QKVD, QKVD);
    convT(proj_W + (size_t)l * DIM * DIM, proj_t + (size_t)l * DIM * DIM, DIM, DIM, DIM);
    convT(mlp_W1 + (size_t)l * DIM * MLPD, mlp1_t + (size_t)l * MLPD * DIM, DIM, MLPD, MLPD);
    convT(mlp_W2 + (size_t)l * MLPD * DIM, mlp2_t + (size_t)l * DIM * MLPD, MLPD, DIM, DIM);
  }

  // --- zero accumulators ---
  zero_f32<<<cdiv(BATCH * RSEG * DIM, 256), 256, 0, stream>>>(psum, BATCH * RSEG * DIM);
  zero_f32<<<cdiv(BATCH * RSEG, 256), 256, 0, stream>>>(cnt, BATCH * RSEG);
  zero_f32<<<cdiv(BATCH * RSEG, 256), 256, 0, stream>>>(pcnt, BATCH * RSEG);
  zero_f32<<<cdiv(BATCH * RSEG, 256), 256, 0, stream>>>(xsum, BATCH * RSEG);
  zero_f32<<<cdiv(BATCH * RSEG, 256), 256, 0, stream>>>(ysum, BATCH * RSEG);

  // --- patch embed ---
  patch_gather<<<cdiv(MPATCH * PKDIM, 256), 256, 0, stream>>>(x, xp16);
  gemm_wmma<false, false, true, false>
      <<<cdiv(gtiles(MPATCH, DIM / 16), 8), 256, 0, stream>>>(
          xp16, pw_t, patch_b, nullptr, emb, nullptr, MPATCH, DIM / 16, DIM, PKDIM);

  // --- superpixel pooling + centroids ---
  pool_accum<<<cdiv(MPATCH * DIM, 256), 256, 0, stream>>>(emb, seg, psum);
  pool_count<<<cdiv(MPATCH, 256), 256, 0, stream>>>(seg, cnt);
  centroid_accum<<<cdiv(BATCH * PIX, 256), 256, 0, stream>>>(seg, pcnt, xsum, ysum);
  assemble_tokens<<<cdiv(MTOK * DIM, 256), 256, 0, stream>>>(
      psum, cnt, pcnt, xsum, ysum, cls_tok, cls_pos, pos_W, pos_b, h);

  // --- transformer layers ---
  int lnBlocks = cdiv(MTOK * 32, 256);
  for (int l = 0; l < DEPTH; ++l) {
    ln_rows<<<lnBlocks, 256, 0, stream>>>(h, DIM, ln1_w + l * DIM, ln1_b + l * DIM, z16, MTOK);
    gemm_wmma<false, false, true, false>
        <<<cdiv(gtiles(MTOK, QKVD / 16), 8), 256, 0, stream>>>(
            z16, qkv_t + (size_t)l * QKVD * DIM, qkv_b + l * QKVD, nullptr,
            qkvbuf, nullptr, MTOK, QKVD / 16, QKVD, DIM);
    banded_attn<<<cdiv(BATCH * HEADS * SEQ, 256), 256, 0, stream>>>(qkvbuf, attn16);
    gemm_wmma<false, true, true, false>
        <<<cdiv(gtiles(MTOK, DIM / 16), 8), 256, 0, stream>>>(
            attn16, proj_t + (size_t)l * DIM * DIM, proj_b + l * DIM, h,
            h, nullptr, MTOK, DIM / 16, DIM, DIM);
    ln_rows<<<lnBlocks, 256, 0, stream>>>(h, DIM, ln2_w + l * DIM, ln2_b + l * DIM, z16, MTOK);
    gemm_wmma<true, false, false, true>
        <<<cdiv(gtiles(MTOK, MLPD / 16), 8), 256, 0, stream>>>(
            z16, mlp1_t + (size_t)l * MLPD * DIM, mlp_b1 + l * MLPD, nullptr,
            nullptr, mid16, MTOK, MLPD / 16, MLPD, DIM);
    gemm_wmma<false, true, true, false>
        <<<cdiv(gtiles(MTOK, DIM / 16), 8), 256, 0, stream>>>(
            mid16, mlp2_t + (size_t)l * DIM * MLPD, mlp_b2 + l * DIM, h,
            h, nullptr, MTOK, DIM / 16, DIM, MLPD);
  }

  // --- final LN on CLS rows + head ---
  ln_rows<<<cdiv(BATCH * 32, 256), 256, 0, stream>>>(
      h, (size_t)SEQ * DIM, norm_w, norm_b, cls16, BATCH);
  gemm_wmma<false, false, true, false>
      <<<cdiv(gtiles(BATCH, HEADPAD / 16), 8), 256, 0, stream>>>(
          cls16, head_t, head_b, nullptr, out, nullptr, BATCH, HEADPAD / 16, CLSN, DIM);
}
